// LongFormer_33646773797417
// MI455X (gfx1250) — matrix operand order repaired and checked by reading, not compiled, not run on hardware
//
#include <hip/hip_runtime.h>
#include <hip/hip_bf16.h>

// Problem constants (match the reference)
#define B_  2
#define S_  4096
#define E_  768
#define H_  12
#define D_  64
#define W_  256
#define BH_ (B_ * H_)
#define BS_ (B_ * S_)

typedef __attribute__((ext_vector_type(16))) _Float16 v16h;
typedef __attribute__((ext_vector_type(8)))  _Float16 v8h;
typedef __attribute__((ext_vector_type(8)))  float    v8f;

static __device__ __forceinline__ v16h join8(v8h lo, v8h hi) {
  return __builtin_shufflevector(lo, hi, 0, 1, 2, 3, 4, 5, 6, 7,
                                 8, 9, 10, 11, 12, 13, 14, 15);
}

static __device__ __forceinline__ v8f wmma16(v16h a, v16h b, v8f c) {
  // v_wmma_f32_16x16x32_f16: (neg_a, A, neg_b, B, c_mod, C, reuse_a, reuse_b)
  return __builtin_amdgcn_wmma_f32_16x16x32_f16(false, a, false, b,
                                                (short)0, c, false, false);
}

// A-fragment (16x32 f16): lane holds row M = lane&15;
// element e -> K = e + 16*half  => one contiguous 16-half run.
static __device__ __forceinline__ v16h load_fragA(const _Float16* base, int ldm,
                                                  int row, int k0, int half) {
  const _Float16* p = base + (size_t)row * ldm + k0 + 16 * half;
  return join8(*(const v8h*)p, *(const v8h*)(p + 8));
}

// B-fragment (32x16 f16): lane holds column N = lane&15; element e -> K = e + 16*half
// => one contiguous 16-half run of row N of B^T (pass B^T row-major).
static __device__ __forceinline__ v16h load_fragB(const _Float16* baseT, int ldm,
                                                  int row, int k0, int half) {
  const _Float16* p = baseT + (size_t)row * ldm + k0 + 16 * half;
  return join8(*(const v8h*)p, *(const v8h*)(p + 8));
}

// Async copy of 16 bytes global -> LDS (GV mode), tracked with ASYNCcnt.
static __device__ __forceinline__ void async_b128(unsigned lds_off,
                                                  const _Float16* g) {
  asm volatile("global_load_async_to_lds_b128 %0, %1, off"
               :: "v"(lds_off), "v"((unsigned long long)(uintptr_t)g)
               : "memory");
}

// ---------------- conversion kernels ----------------

__global__ void cvt_f32_to_f16(const float* __restrict__ src,
                               _Float16* __restrict__ dst, int n) {
  int i = blockIdx.x * blockDim.x + threadIdx.x;
  if (i < n) dst[i] = (_Float16)src[i];
}

__global__ void cvt_transpose_w(const float* __restrict__ Wsrc,
                                _Float16* __restrict__ Wt) {
  int i = blockIdx.x * blockDim.x + threadIdx.x;  // over E*E, coalesced read
  if (i < E_ * E_) {
    int r = i / E_, c = i % E_;
    Wt[(size_t)c * E_ + r] = (_Float16)Wsrc[i];
  }
}

// ---------------- QKV projection GEMM ----------------
// C[BS x E] = X[BS x E] * W[E x E] (+bias). Block = 4 waves; wave = 16 rows x 64 cols.
// W^T tile (64 cols x 32 k) is shared by the 4 waves: staged via double-buffered
// global_load_async_to_lds_b128 so the next chunk's copy overlaps this chunk's WMMAs.
// mode 0: Q -> dstQK [BH,S,D], scaled by 1/sqrt(D)
// mode 1: K -> dstQK [BH,S,D]
// mode 2: V -> dstVt [BH,D,S] (transposed for contiguous P*V B-fragments)
__global__ __launch_bounds__(128) void qkv_gemm(
    const _Float16* __restrict__ X, const _Float16* __restrict__ Wt,
    const float* __restrict__ bias, _Float16* __restrict__ dstQK,
    _Float16* __restrict__ dstVt, int mode) {
  __shared__ _Float16 Wl[2][64 * 32];  // ping-pong W^T tiles, 4KB each

  const int tid  = threadIdx.x;
  const int lane = tid & 31;
  const int wave = tid >> 5;
  const int l16  = lane & 15;
  const int half = lane >> 4;

  const int m0 = (blockIdx.x * 4 + wave) * 16;  // row tile in [0, BS)
  const int h  = blockIdx.y;                    // head == 64-col group (E/64 == H)
  const int n0 = h * 64;

  // Per-thread async-stage assignment: 2048 halves = 128 threads x 2 b128.
  const int wrow = tid >> 1;               // 0..63 : row of W^T tile
  const int wkk  = (tid & 1) * 16;         // 0 or 16 : k-offset (halves)
  const unsigned lds_base =
      (unsigned)(uintptr_t)(&Wl[0][0]);    // ISA: LDS addr = addr[31:0]
  const unsigned lds_off  = (unsigned)((wrow * 32 + wkk) * 2);
  const _Float16* gW = Wt + (size_t)(n0 + wrow) * E_ + wkk;

  // prologue: stage chunk 0 into buffer 0
  async_b128(lds_base + lds_off, gW);
  async_b128(lds_base + lds_off + 16, gW + 8);

  v8f acc[4] = {};
  for (int kc = 0; kc < E_; kc += 32) {
    const int buf = (kc >> 5) & 1;
    asm volatile("s_wait_asynccnt 0" ::: "memory");  // prev prefetch landed
    __syncthreads();                                  // visible to all 4 waves

    // A fragment first: give the global loads maximum slack
    v16h a = load_fragA(X, E_, m0 + l16, kc, half);

    if (kc + 32 < E_) {  // overlap next chunk's copy with this chunk's WMMAs
      const unsigned dst = lds_base + (buf ^ 1) * (64 * 32 * 2) + lds_off;
      async_b128(dst, gW + kc + 32);
      async_b128(dst + 16, gW + kc + 40);
    }

    // Batch all 4 B fragments (8 independent ds_load_b128), then a clean
    // back-to-back WMMA burst with a single wait point.
    v16h bfr[4];
#pragma unroll
    for (int t = 0; t < 4; ++t) {
      const _Float16* wr = &Wl[buf][(t * 16 + l16) * 32 + 16 * half];
      bfr[t] = join8(*(const v8h*)wr, *(const v8h*)(wr + 8));
    }
#pragma unroll
    for (int t = 0; t < 4; ++t) acc[t] = wmma16(a, bfr[t], acc[t]);

    __syncthreads();  // WAR: all waves done reading buf before it is re-staged
  }

  const int bidx = m0 / S_;
  const int s0   = m0 - bidx * S_;
  const int bh   = bidx * H_ + h;
  const float scale = (mode == 0) ? 0.125f : 1.0f;  // 1/sqrt(64)

#pragma unroll
  for (int t = 0; t < 4; ++t) {
    const int d  = t * 16 + l16;
    const float bv = bias[n0 + d];
    if (mode == 2) {
      v8h pack;  // C-layout rows r..r+7 are s-contiguous -> one b128 store
#pragma unroll
      for (int r = 0; r < 8; ++r) pack[r] = (_Float16)(acc[t][r] + bv);
      *(v8h*)(dstVt + ((size_t)bh * D_ + d) * S_ + s0 + 8 * half) = pack;
    } else {
#pragma unroll
      for (int r = 0; r < 8; ++r) {
        const int s = s0 + r + 8 * half;
        dstQK[((size_t)bh * S_ + s) * D_ + d] =
            (_Float16)((acc[t][r] + bv) * scale);
      }
    }
  }
}

// ---------------- sliding-window flash attention ----------------
// One wave per 16-query tile; 17 key blocks of 32 covering [q0-W, q0+15+W].
__global__ __launch_bounds__(128) void attn_kernel(
    const _Float16* __restrict__ Qh,  // [BH,S,D], pre-scaled
    const _Float16* __restrict__ Kh,  // [BH,S,D]
    const _Float16* __restrict__ Vt,  // [BH,D,S]
    float* __restrict__ out) {        // [B,S,E]
  __shared__ _Float16 Plds[4][16 * 32];  // per-wave P staging tile

  const int lane = threadIdx.x & 31;
  const int wave = threadIdx.x >> 5;
  const int l16  = lane & 15;
  const int half = lane >> 4;

  const int wid = blockIdx.x * 4 + wave;
  const int bh  = wid >> 8;  // S/16 == 256 query tiles per bh
  const int q0  = (wid & 255) * 16;

  const _Float16* Qb = Qh + (size_t)bh * S_ * D_;
  const _Float16* Kb = Kh + (size_t)bh * S_ * D_;
  const _Float16* Vb = Vt + (size_t)bh * D_ * S_;

  const v16h qa0 = load_fragA(Qb, D_, q0 + l16, 0, half);
  const v16h qa1 = load_fragA(Qb, D_, q0 + l16, 32, half);

  v8f o[4] = {};
  float mrun[8], lrun[8];
#pragma unroll
  for (int r = 0; r < 8; ++r) { mrun[r] = -3.0e38f; lrun[r] = 0.0f; }

  _Float16* Pw = &Plds[wave][0];

  for (int it = 0; it < 17; ++it) {
    const int kb = q0 - W_ + it * 32;
    if (kb + 31 < 0 || kb >= S_) continue;

    {  // prefetch next key block rows (global_prefetch_b8)
      int pk = kb + 32 + l16;
      pk = pk < 0 ? 0 : (pk > S_ - 1 ? S_ - 1 : pk);
      __builtin_prefetch(Kb + (size_t)pk * D_, 0, 1);
    }

    // ---- scores: two 16x16 tiles, K-dim = D = 2 x 32 ----
    v8f sc[2];
#pragma unroll
    for (int t = 0; t < 2; ++t) {
      const int kt0 = kb + t * 16;
      int kr = kt0 + l16;
      kr = kr < 0 ? 0 : (kr > S_ - 1 ? S_ - 1 : kr);  // clamp; mask fixes value
      v16h kf0 = load_fragB(Kb, D_, kr, 0, half);
      v16h kf1 = load_fragB(Kb, D_, kr, 32, half);
      v8f s = {};
      s = wmma16(qa0, kf0, s);
      s = wmma16(qa1, kf1, s);
#pragma unroll
      for (int r = 0; r < 8; ++r) {  // band + range mask in C layout
        const int m  = q0 + r + 8 * half;
        const int ky = kt0 + l16;
        const bool valid =
            (ky >= 0) && (ky < S_) && (ky - m >= -W_) && (ky - m <= W_);
        if (!valid) s[r] = -3.0e38f;
      }
      sc[t] = s;
    }

    // ---- online softmax (row stats via half-wave shuffles) ----
    float pr0[8], pr1[8], alpha[8];
#pragma unroll
    for (int r = 0; r < 8; ++r) {
      float mx = fmaxf(sc[0][r], sc[1][r]);
#pragma unroll
      for (int sft = 8; sft >= 1; sft >>= 1)
        mx = fmaxf(mx, __shfl_xor(mx, sft, 16));
      const float nm = fmaxf(mrun[r], mx);
      const float p0 = __expf(sc[0][r] - nm);
      const float p1 = __expf(sc[1][r] - nm);
      pr0[r] = p0; pr1[r] = p1;
      float sm = p0 + p1;
#pragma unroll
      for (int sft = 8; sft >= 1; sft >>= 1)
        sm += __shfl_xor(sm, sft, 16);
      alpha[r] = __expf(mrun[r] - nm);
      lrun[r]  = lrun[r] * alpha[r] + sm;
      mrun[r]  = nm;
    }

#pragma unroll
    for (int f = 0; f < 4; ++f)
#pragma unroll
      for (int r = 0; r < 8; ++r) o[f][r] *= alpha[r];

    // ---- re-layout P (C layout -> A fragment) through LDS ----
#pragma unroll
    for (int r = 0; r < 8; ++r) {
      Pw[(r + 8 * half) * 32 + l16]      = (_Float16)pr0[r];
      Pw[(r + 8 * half) * 32 + 16 + l16] = (_Float16)pr1[r];
    }
    asm volatile("s_wait_dscnt 0" ::: "memory");  // cross-lane LDS RAW inside wave

    const _Float16* prow = Pw + l16 * 32;
    const v16h pa =
        join8(*(const v8h*)(prow + 16 * half), *(const v8h*)(prow + 16 * half + 8));

    // ---- O += P x V over 4 d-chunks; Vt gives contiguous B-fragment runs ----
    int kc = kb + 16 * half;
    kc = kc < 0 ? 0 : (kc > S_ - 16 ? S_ - 16 : kc);  // clamp; p==0 where masked
#pragma unroll
    for (int f = 0; f < 4; ++f) {
      const _Float16* vp = Vb + (size_t)(f * 16 + l16) * S_ + kc;
      const v16h vfrag = join8(*(const v8h*)vp, *(const v8h*)(vp + 8));
      o[f] = wmma16(pa, vfrag, o[f]);
    }
  }

  // ---- normalize and scatter back to [B,S,E] ----
  const int b = bh / H_;
  const int h = bh - b * H_;
#pragma unroll
  for (int r = 0; r < 8; ++r) {
    const int s = q0 + r + 8 * half;
    const float inv = 1.0f / lrun[r];
    float* op = out + ((size_t)(b * S_ + s)) * E_ + h * D_;
#pragma unroll
    for (int f = 0; f < 4; ++f) op[f * 16 + l16] = o[f][r] * inv;
  }
}

// ---------------- host launcher ----------------
extern "C" void kernel_launch(void* const* d_in, const int* in_sizes, int n_in,
                              void* d_out, int out_size, void* d_ws, size_t ws_size,
                              hipStream_t stream) {
  (void)in_sizes; (void)n_in; (void)out_size; (void)ws_size;
  const float* hs = (const float*)d_in[0];
  const float* Wq = (const float*)d_in[1];
  const float* bq = (const float*)d_in[2];
  const float* Wk = (const float*)d_in[3];
  const float* bk = (const float*)d_in[4];
  const float* Wv = (const float*)d_in[5];
  const float* bv = (const float*)d_in[6];
  float* out = (float*)d_out;

  char* ws = (char*)d_ws;
  size_t off = 0;
  auto alloc = [&](size_t bytes) -> void* {
    void* p = ws + off;
    off += (bytes + 255) & ~(size_t)255;
    return p;
  };
  _Float16* Xh  = (_Float16*)alloc((size_t)BS_ * E_ * 2);   // hidden f16
  _Float16* Wqt = (_Float16*)alloc((size_t)E_ * E_ * 2);    // W^T f16
  _Float16* Wkt = (_Float16*)alloc((size_t)E_ * E_ * 2);
  _Float16* Wvt = (_Float16*)alloc((size_t)E_ * E_ * 2);
  _Float16* Qh  = (_Float16*)alloc((size_t)BH_ * S_ * D_ * 2);
  _Float16* Kh  = (_Float16*)alloc((size_t)BH_ * S_ * D_ * 2);
  _Float16* Vt  = (_Float16*)alloc((size_t)BH_ * D_ * S_ * 2);

  cvt_f32_to_f16<<<(BS_ * E_) / 256, 256, 0, stream>>>(hs, Xh, BS_ * E_);
  cvt_transpose_w<<<(E_ * E_) / 256, 256, 0, stream>>>(Wq, Wqt);
  cvt_transpose_w<<<(E_ * E_) / 256, 256, 0, stream>>>(Wk, Wkt);
  cvt_transpose_w<<<(E_ * E_) / 256, 256, 0, stream>>>(Wv, Wvt);

  dim3 gg(BS_ / 64, H_, 1);  // 4 waves/block, each 16 rows x 64 cols
  qkv_gemm<<<gg, 128, 0, stream>>>(Xh, Wqt, bq, Qh, Vt, 0);
  qkv_gemm<<<gg, 128, 0, stream>>>(Xh, Wkt, bk, Kh, Vt, 1);
  qkv_gemm<<<gg, 128, 0, stream>>>(Xh, Wvt, bv, Qh, Vt, 2);

  attn_kernel<<<(BH_ * (S_ / 16)) / 4, 128, 0, stream>>>(Qh, Kh, Vt, out);
}
